// MultiHeadAttentionLayer_50431505990065
// MI455X (gfx1250) — compile-verified, matrix operand order checked by reading, not run on hardware
//
#include <hip/hip_runtime.h>
#include <hip/hip_bf16.h>
#include <stdint.h>

typedef __attribute__((ext_vector_type(16))) _Float16 v16h;
typedef __attribute__((ext_vector_type(8)))  _Float16 v8h;
typedef __attribute__((ext_vector_type(8)))  float    v8f;

#define HID   1024
#define HEADS 16
#define HDIM  64
#define BB    4
#define SS    2048

#define BM  128
#define BN  128
#define BK  64
#define LDT 72            // padded LDS row pitch in halfs (144 B, 16B-aligned)
#define KCHUNKS (HID / BK)

union V16U { v16h v; v8h h[2]; };

__device__ __forceinline__ v8f zero8() {
  v8f z = {0.f, 0.f, 0.f, 0.f, 0.f, 0.f, 0.f, 0.f};
  return z;
}

__device__ __forceinline__ v8f wmma_f16(v16h a, v16h b, v8f c) {
  return __builtin_amdgcn_wmma_f32_16x16x32_f16(
      false, a, false, b, (short)0, c, false, false);
}

// WMMA operand fragment from f16 row-major source (global or LDS).
// Lanes 0-15: rows 0-15, K = k0+0..7 then k0+16..23; lanes 16-31: K window +8.
__device__ __forceinline__ v16h frag_from_f16(const _Float16* __restrict__ base,
                                              int ld, int row0, int k0, int lane) {
  const int m = lane & 15, klo = (lane >> 4) << 3;
  const _Float16* p = base + (size_t)(row0 + m) * ld + k0 + klo;
  V16U u;
  u.h[0] = *(const v8h*)(p);
  u.h[1] = *(const v8h*)(p + 16);
  return u.v;
}

// CDNA5 async global->LDS copy (16B per lane), tracked by ASYNCcnt.
__device__ __forceinline__ void async_copy16(const void* g, uint32_t lds_addr) {
  asm volatile("global_load_async_to_lds_b128 %0, %1, off"
               :: "v"(lds_addr), "v"((uint64_t)(uintptr_t)g)
               : "memory");
}
__device__ __forceinline__ void wait_async0() {
  asm volatile("s_wait_asynccnt 0" ::: "memory");
}

// ---------------------------------------------------------------------------
// fp32 -> f16 bulk convert (8 elements / thread, 16B stores)
// ---------------------------------------------------------------------------
__global__ __launch_bounds__(256) void cvt_kernel(const float* __restrict__ src,
                                                  _Float16* __restrict__ dst, int n8) {
  const int i = blockIdx.x * 256 + threadIdx.x;
  if (i >= n8) return;
  const float4 a = *(const float4*)(src + (size_t)i * 8);
  const float4 b = *(const float4*)(src + (size_t)i * 8 + 4);
  v8h h;
  h[0] = (_Float16)a.x; h[1] = (_Float16)a.y; h[2] = (_Float16)a.z; h[3] = (_Float16)a.w;
  h[4] = (_Float16)b.x; h[5] = (_Float16)b.y; h[6] = (_Float16)b.z; h[7] = (_Float16)b.w;
  *(v8h*)(dst + (size_t)i * 8) = h;
}

// ---------------------------------------------------------------------------
// C = A_f16(MxK) @ W_f16(NxK)^T + bias, double-buffered async LDS staging.
// 256 thr = 8 waves; block tile 128x128; wave tile 32x64 (2x4 WMMA frags).
// mode 0: f16 head-split [bh][q][d] (Q,K)
// mode 1: f16 transposed [bh][d][q] (V^T)
// mode 2: f32 row-major (final projection)
// ---------------------------------------------------------------------------
__global__ __launch_bounds__(256) void gemm_kernel(
    const _Float16* __restrict__ A, const _Float16* __restrict__ W,
    const float* __restrict__ bias, _Float16* __restrict__ dst16,
    float* __restrict__ dst32, int mode) {
  extern __shared__ char gsm[];
  _Float16* As = (_Float16*)gsm;                // 2 * BM * LDT halfs
  _Float16* Bs = As + 2 * BM * LDT;             // 2 * BN * LDT halfs

  const int tid  = threadIdx.x;
  const int lane = tid & 31, wid = tid >> 5;
  const int row0 = blockIdx.y * BM;
  const int col0 = blockIdx.x * BN;
  const int wr = (wid & 3) * 32;                // wave row offset in block tile
  const int wc = (wid >> 2) * 64;               // wave col offset in block tile

  const int srow = tid >> 3;                    // staging: 32 rows / pass
  const int sseg = tid & 7;                     // 16B segment within 64-half row

  v8f acc[2][4];
#pragma unroll
  for (int i = 0; i < 2; ++i)
#pragma unroll
    for (int j = 0; j < 4; ++j) acc[i][j] = zero8();

  auto stage = [&](int buf, int kc) {
    const int kbase = kc * BK + sseg * 8;
#pragma unroll
    for (int i = 0; i < 4; ++i) {
      const int r = srow + i * 32;
      const _Float16* ga = A + (size_t)(row0 + r) * HID + kbase;
      async_copy16(ga, (uint32_t)(uintptr_t)(As + buf * BM * LDT + r * LDT + sseg * 8));
      const _Float16* gb = W + (size_t)(col0 + r) * HID + kbase;
      async_copy16(gb, (uint32_t)(uintptr_t)(Bs + buf * BN * LDT + r * LDT + sseg * 8));
    }
  };

  stage(0, 0);
  wait_async0();
  __syncthreads();

  for (int kc = 0; kc < KCHUNKS; ++kc) {
    const int cur = kc & 1;
    if (kc + 1 < KCHUNKS) stage(cur ^ 1, kc + 1);   // prefetch overlaps WMMA

    const _Float16* Ab = As + cur * BM * LDT;
    const _Float16* Bb = Bs + cur * BN * LDT;
#pragma unroll
    for (int ks = 0; ks < BK; ks += 32) {
      const v16h a0 = frag_from_f16(Ab, LDT, wr + 0,  ks, lane);
      const v16h a1 = frag_from_f16(Ab, LDT, wr + 16, ks, lane);
#pragma unroll
      for (int j = 0; j < 4; ++j) {
        const v16h bf = frag_from_f16(Bb, LDT, wc + j * 16, ks, lane);
        acc[0][j] = wmma_f16(a0, bf, acc[0][j]);
        acc[1][j] = wmma_f16(a1, bf, acc[1][j]);
      }
    }
    wait_async0();
    __syncthreads();
  }

  const int n = lane & 15, mb = (lane >> 4) << 3;
#pragma unroll
  for (int i = 0; i < 2; ++i) {
#pragma unroll
    for (int j = 0; j < 4; ++j) {
      const int col = col0 + wc + j * 16 + n;
      const float bvv = bias[col];
      const int h = col >> 6, d = col & 63;
#pragma unroll
      for (int r = 0; r < 8; ++r) {
        const int m = row0 + wr + i * 16 + mb + r;   // global row = b*S + q
        const float v = acc[i][j][r] + bvv;
        if (mode == 2) {
          dst32[(size_t)m * HID + col] = v;
        } else {
          const int b = m >> 11, q = m & (SS - 1);
          const size_t idx = (mode == 1)
              ? ((size_t)((b * HEADS + h) * HDIM + d)) * SS + q
              : ((size_t)((b * HEADS + h) * SS + q)) * HDIM + d;
          dst16[idx] = (_Float16)v;
        }
      }
    }
  }
}

// ---------------------------------------------------------------------------
// Fused attention per (b,h,16-query tile): QK^T -> LDS scores (16x2048 f32),
// masked softmax in LDS, single fp32 attention write, f16 P in LDS, O = P@V.
// ---------------------------------------------------------------------------
__global__ __launch_bounds__(256) void attn_kernel(
    const _Float16* __restrict__ Qh, const _Float16* __restrict__ Kh,
    const _Float16* __restrict__ Vt, const int* __restrict__ mask,
    float* __restrict__ attn, _Float16* __restrict__ Octx) {
  extern __shared__ char smem[];
  float*    Ssc  = (float*)smem;                                 // 16*2048 f32
  _Float16* P    = (_Float16*)(smem + 16 * SS * 4);              // 16*2048 f16
  float*    pred = (float*)(smem + 16 * SS * 4 + 16 * SS * 2);   // [16][16]
  float*    rowv = pred + 256;                                   // rmax[16], rinv[16]

  const int tid = threadIdx.x;
  const int lane = tid & 31, wid = tid >> 5;
  const int bid = blockIdx.x;
  const int qt = bid & (SS / 16 - 1);
  const int bh = bid >> 7;
  const int b = bh >> 4, h = bh & 15;
  const int q0 = qt * 16;

  const _Float16* Qbase = Qh + ((size_t)bh * SS + q0) * HDIM;
  const _Float16* Kbase = Kh + (size_t)bh * SS * HDIM;
  const _Float16* Vbase = Vt + (size_t)bh * HDIM * SS;

  const int n = lane & 15, mb = (lane >> 4) << 3;

  // Phase 1: S = Q K^T / 8, masked
  const v16h aq0 = frag_from_f16(Qbase, HDIM, 0, 0, lane);
  const v16h aq1 = frag_from_f16(Qbase, HDIM, 0, 32, lane);
  for (int kt = wid; kt < SS / 16; kt += 8) {
    v8f acc = zero8();
    acc = wmma_f16(aq0, frag_from_f16(Kbase, HDIM, kt * 16, 0, lane), acc);
    acc = wmma_f16(aq1, frag_from_f16(Kbase, HDIM, kt * 16, 32, lane), acc);
    const int col = kt * 16 + n;
    const int mv = mask[b * SS + col];
#pragma unroll
    for (int r = 0; r < 8; ++r) {
      const float s = acc[r] * 0.125f;
      Ssc[(mb + r) * SS + col] = mv ? s : -1.0e10f;
    }
  }
  __syncthreads();

  // Phase 2: row max
  const int row = tid >> 4, part = tid & 15;
  {
    float mx = -3.0e38f;
    const float* rp = Ssc + row * SS + part * 128;
    for (int c = 0; c < 128; ++c) mx = fmaxf(mx, rp[c]);
    pred[row * 16 + part] = mx;
  }
  __syncthreads();
  if (tid < 16) {
    float mx = pred[tid * 16];
    for (int i = 1; i < 16; ++i) mx = fmaxf(mx, pred[tid * 16 + i]);
    rowv[tid] = mx;
  }
  __syncthreads();

  // Phase 3: row sum of exp
  {
    const float mx = rowv[row];
    float sm = 0.f;
    const float* rp = Ssc + row * SS + part * 128;
    for (int c = 0; c < 128; ++c) sm += __expf(rp[c] - mx);
    pred[row * 16 + part] = sm;
  }
  __syncthreads();
  if (tid < 16) {
    float sm = 0.f;
    for (int i = 0; i < 16; ++i) sm += pred[tid * 16 + i];
    rowv[16 + tid] = 1.0f / sm;
  }
  __syncthreads();

  // Phase 4: probabilities: fp32 once to HBM, f16 to LDS
  float* attn_base = attn + ((size_t)bh * SS + q0) * SS;
  for (int idx = tid; idx < 16 * SS; idx += 256) {
    const int r = idx >> 11;
    const float p = __expf(Ssc[idx] - rowv[r]) * rowv[16 + r];
    attn_base[idx] = p;
    P[idx] = (_Float16)p;
  }
  __syncthreads();

  // Phase 5: O = P @ V (waves 0-3, one 16x16 d-tile each)
  if (wid < 4) {
    const int d0 = wid * 16;
    v8f acc = zero8();
    for (int k0 = 0; k0 < SS; k0 += 32) {
      const v16h ap = frag_from_f16(P, SS, 0, k0, lane);
      const v16h bv = frag_from_f16(Vbase, SS, d0, k0, lane);
      acc = wmma_f16(ap, bv, acc);
    }
#pragma unroll
    for (int r = 0; r < 8; ++r) {
      const int q = q0 + mb + r;
      Octx[((size_t)(b * SS + q)) * HID + h * HDIM + d0 + n] = (_Float16)acc[r];
    }
  }
}

extern "C" void kernel_launch(void* const* d_in, const int* in_sizes, int n_in,
                              void* d_out, int out_size, void* d_ws, size_t ws_size,
                              hipStream_t stream) {
  const float* query  = (const float*)d_in[0];
  const float* key_in = (const float*)d_in[1];
  const float* value  = (const float*)d_in[2];
  const int*   mask   = (const int*)d_in[3];
  const float* Wq = (const float*)d_in[4];
  const float* bq = (const float*)d_in[5];
  const float* Wk = (const float*)d_in[6];
  const float* bk = (const float*)d_in[7];
  const float* Wv = (const float*)d_in[8];
  const float* bv = (const float*)d_in[9];
  const float* Wo = (const float*)d_in[10];
  const float* bo = (const float*)d_in[11];

  float* x_out    = (float*)d_out;
  float* attn_out = x_out + (size_t)BB * SS * HID;

  char* ws = (char*)d_ws;
  _Float16* Xq   = (_Float16*)(ws + ((size_t)0   << 20));  // 16 MB
  _Float16* Xk   = (_Float16*)(ws + ((size_t)16  << 20));  // 16 MB
  _Float16* Xv   = (_Float16*)(ws + ((size_t)32  << 20));  // 16 MB
  _Float16* Wq16 = (_Float16*)(ws + ((size_t)48  << 20));  //  2 MB
  _Float16* Wk16 = (_Float16*)(ws + ((size_t)50  << 20));  //  2 MB
  _Float16* Wv16 = (_Float16*)(ws + ((size_t)52  << 20));  //  2 MB
  _Float16* Wo16 = (_Float16*)(ws + ((size_t)54  << 20));  //  2 MB
  _Float16* Qh   = (_Float16*)(ws + ((size_t)56  << 20));  // 16 MB [bh][q][d]
  _Float16* Kh   = (_Float16*)(ws + ((size_t)72  << 20));  // 16 MB [bh][k][d]
  _Float16* Vt   = (_Float16*)(ws + ((size_t)88  << 20));  // 16 MB [bh][d][k]
  _Float16* Octx = (_Float16*)(ws + ((size_t)104 << 20));  // 16 MB [b*S][HID]

  const dim3 blk(256);
  const int act8 = (BB * SS * HID) / 8;   // 1048576
  const int w8   = (HID * HID) / 8;       // 131072

  cvt_kernel<<<dim3(act8 / 256), blk, 0, stream>>>(query,  Xq, act8);
  cvt_kernel<<<dim3(act8 / 256), blk, 0, stream>>>(key_in, Xk, act8);
  cvt_kernel<<<dim3(act8 / 256), blk, 0, stream>>>(value,  Xv, act8);
  cvt_kernel<<<dim3(w8 / 256),   blk, 0, stream>>>(Wq, Wq16, w8);
  cvt_kernel<<<dim3(w8 / 256),   blk, 0, stream>>>(Wk, Wk16, w8);
  cvt_kernel<<<dim3(w8 / 256),   blk, 0, stream>>>(Wv, Wv16, w8);
  cvt_kernel<<<dim3(w8 / 256),   blk, 0, stream>>>(Wo, Wo16, w8);

  const int gemm_smem = 2 * (BM + BN) * LDT * (int)sizeof(_Float16);  // 73728
  hipFuncSetAttribute(reinterpret_cast<const void*>(gemm_kernel),
                      hipFuncAttributeMaxDynamicSharedMemorySize, gemm_smem);
  const dim3 ggemm(HID / BN, (BB * SS) / BM);  // (8, 64)

  gemm_kernel<<<ggemm, blk, gemm_smem, stream>>>(Xq, Wq16, bq, Qh, nullptr, 0);
  gemm_kernel<<<ggemm, blk, gemm_smem, stream>>>(Xk, Wk16, bk, Kh, nullptr, 0);
  gemm_kernel<<<ggemm, blk, gemm_smem, stream>>>(Xv, Wv16, bv, Vt, nullptr, 1);

  const int attn_smem = 16 * SS * 4 + 16 * SS * 2 + (256 + 32) * 4;
  hipFuncSetAttribute(reinterpret_cast<const void*>(attn_kernel),
                      hipFuncAttributeMaxDynamicSharedMemorySize, attn_smem);
  attn_kernel<<<dim3(BB * HEADS * (SS / 16)), blk, attn_smem, stream>>>(
      Qh, Kh, Vt, mask, attn_out, Octx);

  gemm_kernel<<<ggemm, blk, gemm_smem, stream>>>(Octx, Wo16, bo, nullptr, x_out, 2);
}
